// IdealScoreMachine_62122406969492
// MI455X (gfx1250) — matrix operand order repaired and checked by reading, not compiled, not run on hardware
//
#include <hip/hip_runtime.h>
#include <stdint.h>

#define CHW 3072
#define NIMG 50000
#define BB 64
#define NSEG 16
#define SEG 3136  // NSEG*SEG = 50176 >= 50000, multiple of 32

typedef __attribute__((ext_vector_type(16))) __bf16 v16bf;
typedef __attribute__((ext_vector_type(8)))  float  v8f;
typedef __attribute__((ext_vector_type(4)))  float  f32x4;
typedef __attribute__((ext_vector_type(4)))  unsigned int u32x4;
typedef __attribute__((ext_vector_type(2)))  unsigned int u32x2;

union V16 {
  v16bf  v;
  u32x4  q[2];
  u32x2  d[4];
  __bf16 h[16];
};

__device__ __forceinline__ uint32_t pack_bf16x2(float a, float b) {
  union { __bf16 h[2]; uint32_t u; } p;
  p.h[0] = (__bf16)a; p.h[1] = (__bf16)b;
  return p.u;
}

// ---------------------------------------------------------------- K0: x2 + x->bf16
__global__ __launch_bounds__(256) void k0_prep(const float* __restrict__ x,
                                               float* __restrict__ x2,
                                               __bf16* __restrict__ xbf) {
  __shared__ float red[256];
  const int b = blockIdx.x;
  const int tid = threadIdx.x;
  const float* row = x + (size_t)b * CHW;
  __bf16* orow = xbf + (size_t)b * CHW;
  float s = 0.f;
  for (int k = tid; k < CHW; k += 256) {
    float v = row[k];
    s = fmaf(v, v, s);
    orow[k] = (__bf16)v;
  }
  red[tid] = s;
  __syncthreads();
  for (int off = 128; off > 0; off >>= 1) {
    if (tid < off) red[tid] += red[tid + off];
    __syncthreads();
  }
  if (tid == 0) x2[b] = red[0];
}

// ---------------------------------------------------------------- K1: logits = f(x @ images^T)
__global__ __launch_bounds__(256) void k1_logits(
    const __bf16* __restrict__ xbf, const float* __restrict__ images,
    const float* __restrict__ x2, const float* __restrict__ mu,
    const float* __restrict__ sigma, const int* __restrict__ tptr,
    float* __restrict__ logits) {
  const int lane = threadIdx.x & 31;
  const int wave = threadIdx.x >> 5;
  const int half = lane >> 4;    // 0: lanes 0-15, 1: lanes 16-31
  const int l16  = lane & 15;
  const int n0   = blockIdx.x * 128 + wave * 16;
  const int nrow = n0 + l16;
  const int ncl  = nrow < NIMG ? nrow : NIMG - 1;
  const float* brow = images + (size_t)ncl * CHW;

  v8f acc[4] = {};
  float i2 = 0.f;
  const int aoff = half * 16;   // A-frag K offset within 32-chunk
  const int boff = half * 16;   // B-frag K offset within 32-chunk

  for (int k0 = 0; k0 < CHW; k0 += 32) {
    // B fragment: 16 contiguous fp32 of this image row (K = k0+boff .. +15)
    const f32x4* bp = (const f32x4*)(brow + k0 + boff);
    f32x4 f0 = bp[0], f1 = bp[1], f2 = bp[2], f3 = bp[3];
    i2 = fmaf(f0.x, f0.x, i2); i2 = fmaf(f0.y, f0.y, i2);
    i2 = fmaf(f0.z, f0.z, i2); i2 = fmaf(f0.w, f0.w, i2);
    i2 = fmaf(f1.x, f1.x, i2); i2 = fmaf(f1.y, f1.y, i2);
    i2 = fmaf(f1.z, f1.z, i2); i2 = fmaf(f1.w, f1.w, i2);
    i2 = fmaf(f2.x, f2.x, i2); i2 = fmaf(f2.y, f2.y, i2);
    i2 = fmaf(f2.z, f2.z, i2); i2 = fmaf(f2.w, f2.w, i2);
    i2 = fmaf(f3.x, f3.x, i2); i2 = fmaf(f3.y, f3.y, i2);
    i2 = fmaf(f3.z, f3.z, i2); i2 = fmaf(f3.w, f3.w, i2);
    V16 bf;
    bf.h[0]=(__bf16)f0.x;  bf.h[1]=(__bf16)f0.y;  bf.h[2]=(__bf16)f0.z;  bf.h[3]=(__bf16)f0.w;
    bf.h[4]=(__bf16)f1.x;  bf.h[5]=(__bf16)f1.y;  bf.h[6]=(__bf16)f1.z;  bf.h[7]=(__bf16)f1.w;
    bf.h[8]=(__bf16)f2.x;  bf.h[9]=(__bf16)f2.y;  bf.h[10]=(__bf16)f2.z; bf.h[11]=(__bf16)f2.w;
    bf.h[12]=(__bf16)f3.x; bf.h[13]=(__bf16)f3.y; bf.h[14]=(__bf16)f3.z; bf.h[15]=(__bf16)f3.w;
#pragma unroll
    for (int m = 0; m < 4; ++m) {
      const __bf16* xr = xbf + (size_t)(m * 16 + l16) * CHW + k0;
      V16 af;
      af.q[0] = *(const u32x4*)(xr + aoff);        // K = k0+aoff .. +7
      af.q[1] = *(const u32x4*)(xr + aoff + 8);    // K = k0+aoff+8 .. +7
      acc[m] = __builtin_amdgcn_wmma_f32_16x16x32_bf16(
          false, af.v, false, bf.v, (short)0, acc[m], false, false);
    }
  }

  float i2full = i2 + __shfl_xor(i2, 16);  // combine K-halves (wave32)
  const int t = *tptr;
  const float mu_t = mu[t];
  const float sg   = sigma[t];
  const float inv_var = 1.0f / (sg * sg);
  const float c1 = mu_t * inv_var;
  const float cn = -0.5f * inv_var * mu_t * mu_t * i2full;

  if (nrow < NIMG) {
#pragma unroll
    for (int m = 0; m < 4; ++m) {
#pragma unroll
      for (int r = 0; r < 8; ++r) {
        int b = m * 16 + r + half * 8;
        float lg = fmaf(c1, acc[m][r], cn) - 0.5f * inv_var * x2[b];
        logits[(size_t)b * NIMG + nrow] = lg;
      }
    }
  }
}

// ---------------------------------------------------------------- K2: row softmax -> bf16 weights
__global__ __launch_bounds__(256) void k2_softmax(const float* __restrict__ logits,
                                                  __bf16* __restrict__ wbf) {
  __shared__ float sm[256], ss[256];
  const int b = blockIdx.x, tid = threadIdx.x;
  const float* row = logits + (size_t)b * NIMG;
  float m = -3.0e38f, s = 0.f;
  for (int n = tid; n < NIMG; n += 256) {
    float v = row[n];
    if (v > m) { s = s * __expf(m - v) + 1.f; m = v; }
    else       { s += __expf(v - m); }
  }
  sm[tid] = m; ss[tid] = s;
  __syncthreads();
  for (int off = 128; off > 0; off >>= 1) {
    if (tid < off) {
      float m1 = sm[tid], s1 = ss[tid];
      float m2 = sm[tid + off], s2 = ss[tid + off];
      float M = m1 > m2 ? m1 : m2;
      sm[tid] = M;
      ss[tid] = s1 * __expf(m1 - M) + s2 * __expf(m2 - M);
    }
    __syncthreads();
  }
  const float M = sm[0];
  const float invS = 1.f / ss[0];
  __bf16* wrow = wbf + (size_t)b * NIMG;
  for (int n = tid; n < NIMG; n += 256) {
    wrow[n] = (__bf16)(__expf(row[n] - M) * invS);
  }
}

// ---------------------------------------------------------------- K3: xbar partials = w @ images
__global__ __launch_bounds__(256) void k3_xbar(const float* __restrict__ images,
                                               const __bf16* __restrict__ wbf,
                                               float* __restrict__ parts) {
  __shared__ __bf16 T[128 * 36];  // T[kp_local][n_local], padded stride 36
  const int tid  = threadIdx.x;
  const int lane = tid & 31, wave = tid >> 5;
  const int half = lane >> 4, l16 = lane & 15;
  const int kp0  = blockIdx.x * 128;
  const int nbeg = blockIdx.y * SEG;
  const int nend = min(nbeg + SEG, NIMG);

  const int kpg   = tid & 15;   // 16 groups of 8 columns
  const int npair = tid >> 4;   // rows 2*npair, 2*npair+1

  v8f acc[4] = {};

  for (int nb = nbeg; nb < nend; nb += 32) {
    __syncthreads();
    {
      int r0 = nb + 2 * npair, r1 = r0 + 1;
      int rc0 = r0 < NIMG ? r0 : NIMG - 1;
      int rc1 = r1 < NIMG ? r1 : NIMG - 1;
      const float* p0 = images + (size_t)rc0 * CHW + kp0 + kpg * 8;
      const float* p1 = images + (size_t)rc1 * CHW + kp0 + kpg * 8;
      f32x4 a0 = *(const f32x4*)(p0), a1 = *(const f32x4*)(p0 + 4);
      f32x4 b0 = *(const f32x4*)(p1), b1 = *(const f32x4*)(p1 + 4);
      float va[8] = {a0.x,a0.y,a0.z,a0.w,a1.x,a1.y,a1.z,a1.w};
      float vb[8] = {b0.x,b0.y,b0.z,b0.w,b1.x,b1.y,b1.z,b1.w};
#pragma unroll
      for (int j = 0; j < 8; ++j) {
        int col = kpg * 8 + j;
        *(uint32_t*)&T[col * 36 + 2 * npair] = pack_bf16x2(va[j], vb[j]);
      }
    }
    __syncthreads();
    V16 bf;  // B frag: column kp = kp0+wave*16+l16, K = n_local half*16..+15
    {
      const u32x2* tp = (const u32x2*)&T[(wave * 16 + l16) * 36 + half * 16];
      bf.d[0] = tp[0]; bf.d[1] = tp[1]; bf.d[2] = tp[2]; bf.d[3] = tp[3];
    }
#pragma unroll
    for (int m = 0; m < 4; ++m) {
      const __bf16* wr = wbf + (size_t)(m * 16 + l16) * NIMG;
      int nA0 = nb + half * 16;
      int nA1 = nb + half * 16 + 8;
      u32x4 z = {};
      V16 af;
      af.q[0] = (nA0 < nend) ? *(const u32x4*)(wr + nA0) : z;
      af.q[1] = (nA1 < nend) ? *(const u32x4*)(wr + nA1) : z;
      acc[m] = __builtin_amdgcn_wmma_f32_16x16x32_bf16(
          false, af.v, false, bf.v, (short)0, acc[m], false, false);
    }
  }

  float* pp = parts + (size_t)blockIdx.y * (BB * CHW);
#pragma unroll
  for (int m = 0; m < 4; ++m) {
#pragma unroll
    for (int r = 0; r < 8; ++r) {
      int b = m * 16 + r + half * 8;
      pp[(size_t)b * CHW + kp0 + wave * 16 + l16] = acc[m][r];
    }
  }
}

// ---------------------------------------------------------------- K4: reduce + epilogue
__global__ __launch_bounds__(256) void k4_out(const float* __restrict__ x,
                                              const float* __restrict__ parts,
                                              const float* __restrict__ mu,
                                              const float* __restrict__ sigma,
                                              const int* __restrict__ tptr,
                                              float* __restrict__ out) {
  int i = blockIdx.x * 256 + threadIdx.x;
  if (i >= BB * CHW) return;
  float s = 0.f;
#pragma unroll
  for (int g = 0; g < NSEG; ++g) s += parts[(size_t)g * (BB * CHW) + i];
  const int t = *tptr;
  const float mu_t = mu[t];
  const float sg   = sigma[t];
  const float inv_var = 1.0f / (sg * sg);
  out[i] = (mu_t * s - x[i]) * inv_var;
}

// ---------------------------------------------------------------- launch
extern "C" void kernel_launch(void* const* d_in, const int* in_sizes, int n_in,
                              void* d_out, int out_size, void* d_ws, size_t ws_size,
                              hipStream_t stream) {
  const float* x      = (const float*)d_in[0];
  const float* images = (const float*)d_in[1];
  const float* mu     = (const float*)d_in[2];
  const float* sigma  = (const float*)d_in[3];
  const int*   t      = (const int*)d_in[4];
  float* out = (float*)d_out;

  char* ws = (char*)d_ws;
  size_t off = 0;
  float* logits = (float*)(ws + off); off += (size_t)BB * NIMG * sizeof(float);
  float* x2     = (float*)(ws + off); off += 64 * sizeof(float);
  off = (off + 255) & ~(size_t)255;
  float* parts  = (float*)(ws + off); off += (size_t)NSEG * BB * CHW * sizeof(float);
  __bf16* xbf   = (__bf16*)(ws + off); off += (size_t)BB * CHW * sizeof(__bf16);
  off = (off + 255) & ~(size_t)255;
  __bf16* wbf   = (__bf16*)(ws + off);

  k0_prep   <<<BB, 256, 0, stream>>>(x, x2, xbf);
  k1_logits <<<(NIMG + 127) / 128, 256, 0, stream>>>(xbf, images, x2, mu, sigma, t, logits);
  k2_softmax<<<BB, 256, 0, stream>>>(logits, wbf);
  k3_xbar   <<<dim3(CHW / 128, NSEG), 256, 0, stream>>>(images, wbf, parts);
  k4_out    <<<(BB * CHW + 255) / 256, 256, 0, stream>>>(x, parts, mu, sigma, t, out);
}
